// SparseVGGStage_50354196578893
// MI455X (gfx1250) — matrix operand order repaired and checked by reading, not compiled, not run on hardware
//
#include <hip/hip_runtime.h>

// ---------------- geometry constants (match reference) ----------------
#define NB   4
#define NH   512
#define NW   512
#define NPTS 262144                  // active sites
#define NP2  (NB*(NH/2)*(NW/2))      // pooled dense grid = 262144
#define COUT 128

// ---------------- CDNA5 WMMA types ----------------
typedef __attribute__((ext_vector_type(8)))  __bf16 v8bf;
typedef __attribute__((ext_vector_type(16))) __bf16 v16bf;
typedef __attribute__((ext_vector_type(8)))  float  v8f;
typedef __attribute__((ext_vector_type(4)))  int    v4i;

#define AS1 __attribute__((address_space(1)))
#define AS3 __attribute__((address_space(3)))

#if defined(__HIP_DEVICE_COMPILE__) && \
    __has_builtin(__builtin_amdgcn_global_load_async_to_lds_b128)
#define SVG_ASYNC 1
#else
#define SVG_ASYNC 0
#endif

__device__ __forceinline__ void svg_wait_async() {
#if SVG_ASYNC
#if __has_builtin(__builtin_amdgcn_s_wait_asynccnt)
  __builtin_amdgcn_s_wait_asynccnt(0);
#else
  asm volatile("s_wait_asynccnt 0x0" ::: "memory");
#endif
#endif
}

__device__ __forceinline__ void svg_async_b128(const unsigned short* gsrc,
                                               unsigned short* ldst) {
#if SVG_ASYNC
  __builtin_amdgcn_global_load_async_to_lds_b128(
      (AS1 v4i*)gsrc, (AS3 v4i*)ldst, 0, 0);
#else
  *(uint4*)ldst = *(const uint4*)gsrc;
#endif
}

__device__ __forceinline__ unsigned short svg_f2bf(float f) {
  unsigned u = __float_as_uint(f);
  unsigned r = u + 0x7FFFu + ((u >> 16) & 1u);   // round-to-nearest-even
  return (unsigned short)(r >> 16);
}

// ---------------- utility kernels ----------------
__global__ void svg_fill_i32(int* __restrict__ p, int v, long long n) {
  long long i = (long long)blockIdx.x * blockDim.x + threadIdx.x;
  long long st = (long long)gridDim.x * blockDim.x;
  for (; i < n; i += st) p[i] = v;
}

__global__ void svg_cvt_bf16(const float* __restrict__ x,
                             unsigned short* __restrict__ y, long long n) {
  long long i = (long long)blockIdx.x * blockDim.x + threadIdx.x;
  long long st = (long long)gridDim.x * blockDim.x;
  for (; i < n; i += st) y[i] = svg_f2bf(x[i]);
}

// W [9][cin][128] fp32  ->  wT [9][cin/32][128][32] bf16 (k-innermost)
__global__ void svg_wtrans(const float* __restrict__ Wsrc,
                           unsigned short* __restrict__ wT, int cin) {
  long long total = (long long)9 * cin * 128;
  long long i = (long long)blockIdx.x * blockDim.x + threadIdx.x;
  long long st = (long long)gridDim.x * blockDim.x;
  for (; i < total; i += st) {
    int kk  = (int)(i / (cin * 128));
    int rem = (int)(i % (cin * 128));
    int kc  = rem >> 12;               // / (128*32)
    int r2  = rem & 4095;
    int n   = r2 >> 5;
    int k2  = r2 & 31;
    int k   = kc * 32 + k2;
    wT[i] = svg_f2bf(Wsrc[((size_t)kk * cin + k) * 128 + n]);
  }
}

__global__ void svg_scatter_grid(const int* __restrict__ coords,
                                 int* __restrict__ gmap) {
  int i = blockIdx.x * blockDim.x + threadIdx.x;
  if (i >= NPTS) return;
  int b = coords[3*i], y = coords[3*i+1], x = coords[3*i+2];
  gmap[(b * NH + y) * NW + x] = i;
}

// ------- gather + WMMA conv: 64 rows x 128 out-channels per block -------
// Per 3x3 offset: stage gathered A rows (64 x CIN bf16) and the whole
// transposed weight slab (CIN x 128 bf16) into LDS via async global->LDS,
// then run the entire K loop barrier-free on v_wmma_f32_16x16x32_bf16.
template<int CIN>
__global__ __launch_bounds__(256)
void svg_conv(const unsigned short* __restrict__ fin,   // [NPTS, CIN] bf16
              const unsigned short* __restrict__ wgt,   // [9][CIN/32][128][32] bf16
              const float* __restrict__ bias,           // [128]
              const int* __restrict__ coords,           // [NPTS,3]
              const int* __restrict__ gmap,             // [NB*NH*NW]
              float* __restrict__ fout)                 // [NPTS,128] fp32
{
  __shared__ __align__(16) unsigned short sA[64 * CIN];   // gathered rows
  __shared__ __align__(16) unsigned short sW[CIN * 128];  // weights [kc][n][k]
  __shared__ int sSrc[64];

  const int tid  = threadIdx.x;
  const int lane = tid & 31;
  const int wv   = tid >> 5;
  const int mw   = wv >> 1;        // 0..3 : 16-row strip
  const int nw   = wv & 1;         // 0..1 : 64-col strip
  const int half = lane >> 4;      // wave32 half
  const int l15  = lane & 15;
  const int r0   = blockIdx.x * 64;

  v8f acc[4];
#pragma unroll
  for (int t = 0; t < 4; ++t)
#pragma unroll
    for (int i = 0; i < 8; ++i) acc[t][i] = 0.0f;

  for (int kk = 0; kk < 9; ++kk) {
    if (tid < 64) {                               // rulebook lookup
      int r = r0 + tid;
      int cb = coords[3*r], cy = coords[3*r+1], cx = coords[3*r+2];
      int ny = cy - (kk / 3 - 1);
      int nx = cx - (kk % 3 - 1);
      int s = -1;
      if (ny >= 0 && ny < NH && nx >= 0 && nx < NW)
        s = gmap[(cb * NH + ny) * NW + nx];
      sSrc[tid] = s;
    }
    __syncthreads();   // sSrc ready; previous kk's compute done with sA/sW

    { // stage gathered A rows: 4 threads per row, CIN/4 halfs each
      int row  = tid >> 2;
      int koff = (tid & 3) * (CIN / 4);
      int s = sSrc[row];
      unsigned short* dst = sA + row * CIN + koff;
      if (s >= 0) {
        const unsigned short* src = fin + (size_t)s * CIN + koff;
#pragma unroll
        for (int q = 0; q < CIN / 32; ++q)
          svg_async_b128(src + q * 8, dst + q * 8);
      } else {
        uint4 z = make_uint4(0u, 0u, 0u, 0u);
#pragma unroll
        for (int q = 0; q < CIN / 32; ++q)
          *(uint4*)(dst + q * 8) = z;
      }
    }
    { // stage weight slab for this offset: contiguous CIN*128 halfs
      const unsigned short* src = wgt + (size_t)kk * CIN * 128 + tid * (CIN / 2);
      unsigned short* dst = sW + tid * (CIN / 2);
#pragma unroll
      for (int q = 0; q < CIN / 16; ++q)
        svg_async_b128(src + q * 8, dst + q * 8);
    }
    svg_wait_async();
    __syncthreads();

#pragma unroll
    for (int kc = 0; kc < CIN / 32; ++kc) {
      // A fragment: lanes 0-15 row=l15 K={0..7,16..23}; 16-31 K={8..15,24..31}
      const v8bf* pa =
          (const v8bf*)(sA + (mw * 16 + l15) * CIN + kc * 32 + (half ? 8 : 0));
      v16bf afrag = __builtin_shufflevector(pa[0], pa[2],
          0,1,2,3,4,5,6,7,8,9,10,11,12,13,14,15);
#pragma unroll
      for (int t = 0; t < 4; ++t) {
        // B fragment: lanes 0-15 col=l15 K={0..7,16..23}; 16-31 K={8..15,24..31}
        const v8bf* pb = (const v8bf*)(sW +
            ((size_t)kc * 128 + nw * 64 + t * 16 + l15) * 32 + (half ? 8 : 0));
        v16bf bfrag = __builtin_shufflevector(pb[0], pb[2],
            0,1,2,3,4,5,6,7,8,9,10,11,12,13,14,15);
        acc[t] = __builtin_amdgcn_wmma_f32_16x16x32_bf16(
            false, afrag, false, bfrag, (short)0, acc[t], false, false);
      }
    }
    __syncthreads();   // all waves done reading sA/sW before restaging
  }

  // epilogue: C layout — VGPR i: lanes 0-15 -> M=i, lanes 16-31 -> M=i+8
#pragma unroll
  for (int t = 0; t < 4; ++t) {
    int bcol = nw * 64 + t * 16 + l15;
    float bs = bias[bcol];
#pragma unroll
    for (int i = 0; i < 8; ++i) {
      int mrow = r0 + mw * 16 + (half ? i + 8 : i);
      fout[(size_t)mrow * COUT + bcol] = acc[t][i] + bs;
    }
  }
}

// ---------------- deterministic BN statistics ----------------
__global__ __launch_bounds__(128)
void svg_bnstats(const float* __restrict__ x, float* __restrict__ part, int nrows) {
  int c = threadIdx.x;                       // 0..127
  float s = 0.f, q = 0.f;
  for (int r = blockIdx.x; r < nrows; r += gridDim.x) {
    float v = x[(size_t)r * COUT + c];
    s += v; q += v * v;
  }
  part[(size_t)blockIdx.x * 256 + c]       = s;
  part[(size_t)blockIdx.x * 256 + 128 + c] = q;
}

__global__ __launch_bounds__(128)
void svg_bnfinal(const float* __restrict__ part, const float* __restrict__ g,
                 const float* __restrict__ be, float* __restrict__ ss,
                 int nblocks, float n) {
  int t = threadIdx.x;
  float s = 0.f, q = 0.f;
  for (int b = 0; b < nblocks; ++b) {        // fixed order -> deterministic
    s += part[(size_t)b * 256 + t];
    q += part[(size_t)b * 256 + 128 + t];
  }
  float mu  = s / n;
  float var = q / n - mu * mu;
  float sc  = g[t] * rsqrtf(var + 1e-5f);
  ss[t]       = sc;
  ss[128 + t] = be[t] - mu * sc;
}

__global__ void svg_bnrelu_bf(const float* __restrict__ x,
                              const float* __restrict__ ss,
                              unsigned short* __restrict__ y, long long total) {
  long long i = (long long)blockIdx.x * blockDim.x + threadIdx.x;
  long long st = (long long)gridDim.x * blockDim.x;
  for (; i < total; i += st) {
    int c = (int)(i & 127);
    float v = x[i] * ss[c] + ss[128 + c];
    y[i] = svg_f2bf(v > 0.f ? v : 0.f);
  }
}

// ---------------- sparse 2x2 max pool: occupancy, scan, scatter ----------------
__global__ void svg_occ(const int* __restrict__ coords, int* __restrict__ occ) {
  int i = blockIdx.x * blockDim.x + threadIdx.x;
  if (i >= NPTS) return;
  int b = coords[3*i], y = coords[3*i+1], x = coords[3*i+2];
  occ[((b * (NH/2)) + (y >> 1)) * (NW/2) + (x >> 1)] = 1;
}

__global__ __launch_bounds__(256)
void svg_scan1(const int* __restrict__ occ, int* __restrict__ scn,
               int* __restrict__ bsum) {
  __shared__ int tmp[256];
  int t = threadIdx.x, i = blockIdx.x * 256 + t;
  int v = occ[i];
  tmp[t] = v; __syncthreads();
  for (int off = 1; off < 256; off <<= 1) {
    int u = (t >= off) ? tmp[t - off] : 0;
    __syncthreads();
    tmp[t] += u;
    __syncthreads();
  }
  scn[i] = tmp[t] - v;                       // block-local exclusive
  if (t == 255) bsum[blockIdx.x] = tmp[t];
}

__global__ __launch_bounds__(1024)
void svg_scan2(const int* __restrict__ bsum, int* __restrict__ boff) {
  __shared__ int tmp[1024];
  int t = threadIdx.x;
  int v = bsum[t];
  tmp[t] = v; __syncthreads();
  for (int off = 1; off < 1024; off <<= 1) {
    int u = (t >= off) ? tmp[t - off] : 0;
    __syncthreads();
    tmp[t] += u;
    __syncthreads();
  }
  boff[t] = tmp[t] - v;                      // exclusive
}

__global__ __launch_bounds__(256)
void svg_scan3_uniq(int* __restrict__ scn, const int* __restrict__ boff,
                    const int* __restrict__ occ, int* __restrict__ uniq) {
  int i = blockIdx.x * 256 + threadIdx.x;
  int v = scn[i] + boff[blockIdx.x];
  scn[i] = v;
  if (occ[i]) {                              // lexicographic (b, y//2, x//2) order
    int b = i >> 16, rem = i & 0xFFFF;
    uniq[3*v + 0] = b;
    uniq[3*v + 1] = rem >> 8;
    uniq[3*v + 2] = rem & 0xFF;
  }
}

__global__ void svg_pool(const float* __restrict__ x, const float* __restrict__ ss,
                         const int* __restrict__ coords, const int* __restrict__ scn,
                         float* __restrict__ pooled, long long total) {
  long long idx = (long long)blockIdx.x * blockDim.x + threadIdx.x;
  long long st  = (long long)gridDim.x * blockDim.x;
  for (; idx < total; idx += st) {
    int i = (int)(idx >> 7), c = (int)(idx & 127);
    float v = x[idx] * ss[c] + ss[128 + c];
    v = v > 0.f ? v : 0.f;                   // relu -> non-negative
    int b = coords[3*i], y = coords[3*i+1], xx = coords[3*i+2];
    int seg = scn[((b * (NH/2)) + (y >> 1)) * (NW/2) + (xx >> 1)];
    // non-negative floats order like their int bit patterns
    atomicMax((int*)&pooled[(size_t)seg * COUT + c], __float_as_int(v));
  }
}

// ---------------- host-side orchestration ----------------
extern "C" void kernel_launch(void* const* d_in, const int* in_sizes, int n_in,
                              void* d_out, int out_size, void* d_ws, size_t ws_size,
                              hipStream_t stream) {
  const int*   coords = (const int*)  d_in[0];
  const float* feats  = (const float*)d_in[1];
  const float* W1     = (const float*)d_in[2];
  const float* b1     = (const float*)d_in[3];
  const float* g1     = (const float*)d_in[4];
  const float* be1    = (const float*)d_in[5];
  const float* W2     = (const float*)d_in[6];
  const float* b2     = (const float*)d_in[7];
  const float* g2     = (const float*)d_in[8];
  const float* be2    = (const float*)d_in[9];

  // carve workspace (256B aligned chunks)
  char* p = (char*)d_ws;
  auto carve = [&](size_t bytes) {
    void* r = (void*)p;
    p += (bytes + 255) & ~(size_t)255;
    return r;
  };
  int*   gmap    = (int*)carve((size_t)NB * NH * NW * 4);     // 4 MB
  int*   occ     = (int*)carve((size_t)NP2 * 4);              // 1 MB
  int*   scn     = (int*)carve((size_t)NP2 * 4);              // 1 MB
  int*   bsum    = (int*)carve(1024 * 4);
  int*   boff    = (int*)carve(1024 * 4);
  float* part    = (float*)carve((size_t)512 * 256 * 4);      // BN partials
  float* ssA     = (float*)carve(256 * 4);
  float* ssB     = (float*)carve(256 * 4);
  unsigned short* featsbf = (unsigned short*)carve((size_t)NPTS * 64 * 2);
  unsigned short* wT1     = (unsigned short*)carve((size_t)9 * 64 * 128 * 2);
  unsigned short* wT2     = (unsigned short*)carve((size_t)9 * 128 * 128 * 2);
  unsigned short* f1bf    = (unsigned short*)carve((size_t)NPTS * 128 * 2);
  float* out1    = (float*)carve((size_t)NPTS * 128 * 4);     // reused for conv2 out
  (void)ws_size; (void)n_in; (void)in_sizes;

  const int M = out_size / (3 + COUT);       // unique pooled sites (host-known)
  int*   uniq   = (int*)d_out;
  float* pooled = (float*)d_out + (size_t)3 * M;

  // init buffers touched by scatter/atomics (deterministic every call)
  svg_fill_i32<<<4096, 256, 0, stream>>>(gmap, -1, (long long)NB * NH * NW);
  svg_fill_i32<<<1024, 256, 0, stream>>>(occ, 0, (long long)NP2);
  svg_fill_i32<<<4096, 256, 0, stream>>>((int*)d_out, 0, (long long)out_size);

  // rulebook + bf16 conversion + weight pre-transpose ([kk][kc][n][k])
  svg_scatter_grid<<<NPTS / 256, 256, 0, stream>>>(coords, gmap);
  svg_cvt_bf16<<<8192, 256, 0, stream>>>(feats, featsbf, (long long)NPTS * 64);
  svg_wtrans<<<288, 256, 0, stream>>>(W1, wT1, 64);
  svg_wtrans<<<576, 256, 0, stream>>>(W2, wT2, 128);

  // layer 1: gather-conv (WMMA) -> BN stats -> BN+ReLU -> bf16
  svg_conv<64><<<NPTS / 64, 256, 0, stream>>>(featsbf, wT1, b1, coords, gmap, out1);
  svg_bnstats<<<512, 128, 0, stream>>>(out1, part, NPTS);
  svg_bnfinal<<<1, 128, 0, stream>>>(part, g1, be1, ssA, 512, (float)NPTS);
  svg_bnrelu_bf<<<16384, 256, 0, stream>>>(out1, ssA, f1bf, (long long)NPTS * 128);

  // layer 2: gather-conv (WMMA) -> BN stats
  svg_conv<128><<<NPTS / 64, 256, 0, stream>>>(f1bf, wT2, b2, coords, gmap, out1);
  svg_bnstats<<<512, 128, 0, stream>>>(out1, part, NPTS);
  svg_bnfinal<<<1, 128, 0, stream>>>(part, g2, be2, ssB, 512, (float)NPTS);

  // sparse 2x2 max-pool: occupancy -> exclusive scan -> uniq coords -> seg max
  svg_occ<<<NPTS / 256, 256, 0, stream>>>(coords, occ);
  svg_scan1<<<NP2 / 256, 256, 0, stream>>>(occ, scn, bsum);
  svg_scan2<<<1, 1024, 0, stream>>>(bsum, boff);
  svg_scan3_uniq<<<NP2 / 256, 256, 0, stream>>>(scn, boff, occ, uniq);
  svg_pool<<<16384, 256, 0, stream>>>(out1, ssB, coords, scn, pooled,
                                      (long long)NPTS * 128);
}
